// PlainGAT_39805756900149
// MI455X (gfx1250) — matrix operand order repaired and checked by reading, not compiled, not run on hardware
//
#include <hip/hip_runtime.h>
#include <hip/hip_bf16.h>
#include <math.h>

#define N_NODES 50000
#define N_EDGES 1600000
#define MODEL_DIM 128
#define NUM_HEADS 8
#define D_V 16
#define GAT_EPS 1e-12f

typedef __attribute__((ext_vector_type(16))) _Float16 v16h;
typedef __attribute__((ext_vector_type(8)))  float    v8f;

// ---------------------------------------------------------------------------
// k0: zero denom/attn_sum, reset global-max cell
// ---------------------------------------------------------------------------
__global__ void gat_zero_kernel(float* denom, float* asum, unsigned* gmax, int n) {
    int i = blockIdx.x * blockDim.x + threadIdx.x;
    if (i < n) { denom[i] = 0.0f; asum[i] = 0.0f; }
    if (i == 0) *gmax = 0u;   // below encode(-inf)=0x007FFFFF -> safe identity
}

// ---------------------------------------------------------------------------
// k1: WMMA GEMM.  value = inp @ Wv^T (-> ws), out = inp @ Wself^T + bias (-> d_out)
// One wave handles a 16-row tile of nodes. wave32.
// ---------------------------------------------------------------------------
__global__ __launch_bounds__(256) void gat_gemm_wmma(
        const float* __restrict__ inp, const float* __restrict__ Wv,
        const float* __restrict__ Wself, const float* __restrict__ bias,
        float* __restrict__ value, float* __restrict__ out)
{
    const int lane = threadIdx.x & 31;
    const int wave = threadIdx.x >> 5;
    const int tile = blockIdx.x * 8 + wave;
    if (tile * 16 >= N_NODES) return;
    const int row0    = tile * 16;
    const int halfsel = lane >> 4;   // lane group 0..15 vs 16..31
    const int lrow    = lane & 15;

    // A fragments (16x32 f16 per K-block). Contiguous k order used here (B matches):
    //   element e of v16h on lane l holds K = e + 16*(l>>4)
    v16h afrag[4];
    const float* arow = inp + (size_t)(row0 + lrow) * MODEL_DIM;
#pragma unroll
    for (int kb = 0; kb < 4; ++kb) {
        const int base = kb * 32 + halfsel * 16;
#pragma unroll
        for (int e = 0; e < 16; ++e) {
            const int K = base + e;
            afrag[kb][e] = (_Float16)arow[K];
        }
    }

    // 8 column tiles of 16 (one per head for the value matrix)
    for (int n = 0; n < 8; ++n) {
        const int col0 = n * 16;
        // B column N = W row (col0+lrow), contiguous in memory.
        // B layout: element e on lane l holds K = e + 16*(l>>4)
        const float* wvrow = Wv    + (size_t)(col0 + lrow) * MODEL_DIM;
        const float* wsrow = Wself + (size_t)(col0 + lrow) * MODEL_DIM;
        v8f cv = {}; v8f cs = {};
#pragma unroll
        for (int kb = 0; kb < 4; ++kb) {
            const int kbase = kb * 32 + halfsel * 16;
            v16h bv, bs;
#pragma unroll
            for (int e = 0; e < 16; ++e) {
                bv[e] = (_Float16)wvrow[kbase + e];
                bs[e] = (_Float16)wsrow[kbase + e];
            }
            cv = __builtin_amdgcn_wmma_f32_16x16x32_f16(false, afrag[kb], false, bv,
                                                        (short)0, cv, false, false);
            cs = __builtin_amdgcn_wmma_f32_16x16x32_f16(false, afrag[kb], false, bs,
                                                        (short)0, cs, false, false);
        }
        // C/D layout: VGPR r on lane l -> M = r + 8*(l>>4), N = l&15
        const float b = bias[col0 + lrow];
#pragma unroll
        for (int r = 0; r < 8; ++r) {
            const int row = row0 + r + 8 * halfsel;
            value[(size_t)row * MODEL_DIM + col0 + lrow] = cv[r];
            out  [(size_t)row * MODEL_DIM + col0 + lrow] = cs[r] + b;
        }
    }
}

// ---------------------------------------------------------------------------
// k2: per-head source/target projections: vsrc/vtgt[n][h] = <value[n,h,:], w>
// ---------------------------------------------------------------------------
__global__ void gat_proj_kernel(const float* __restrict__ value,
                                const float* __restrict__ w_src,
                                const float* __restrict__ w_tgt,
                                float* __restrict__ vsrc, float* __restrict__ vtgt)
{
    int tid = blockIdx.x * blockDim.x + threadIdx.x;   // node*8 + head
    if (tid >= N_NODES * NUM_HEADS) return;
    const int n = tid >> 3, h = tid & 7;
    const float* v = value + (size_t)n * MODEL_DIM + h * D_V;
    const float* a = w_src + h * D_V;
    const float* b = w_tgt + h * D_V;
    float s = 0.0f, t = 0.0f;
#pragma unroll
    for (int d = 0; d < D_V; ++d) { s += v[d] * a[d]; t += v[d] * b[d]; }
    vsrc[tid] = s; vtgt[tid] = t;
}

// order-preserving float->uint encoding for atomic max over signed floats
__device__ __forceinline__ unsigned enc_f32(float f) {
    unsigned u = __float_as_uint(f);
    return (u & 0x80000000u) ? ~u : (u | 0x80000000u);
}
__device__ __forceinline__ float dec_f32(unsigned u) {
    return __uint_as_float((u & 0x80000000u) ? (u & 0x7FFFFFFFu) : ~u);
}

// ---------------------------------------------------------------------------
// k3: raw leaky-relu scores (-> attn scratch region of d_out) + global max
// ---------------------------------------------------------------------------
__global__ void gat_score_kernel(const int* __restrict__ src, const int* __restrict__ tgt,
                                 const float* __restrict__ vsrc, const float* __restrict__ vtgt,
                                 float* __restrict__ attn, unsigned* __restrict__ gmax)
{
    __shared__ unsigned red[256];
    int tid = blockIdx.x * blockDim.x + threadIdx.x;   // edge*8 + head
    unsigned mine = 0u;
    if (tid < N_EDGES * NUM_HEADS) {
        const int e = tid >> 3, h = tid & 7;
        float s = vsrc[src[e] * NUM_HEADS + h] + vtgt[tgt[e] * NUM_HEADS + h];
        s = (s >= 0.0f) ? s : 0.2f * s;                // leaky_relu(0.2)
        attn[tid] = s;
        mine = enc_f32(s);
    }
    red[threadIdx.x] = mine;
    __syncthreads();
    for (int off = 128; off > 0; off >>= 1) {
        if (threadIdx.x < off) {
            unsigned o = red[threadIdx.x + off];
            if (o > red[threadIdx.x]) red[threadIdx.x] = o;
        }
        __syncthreads();
    }
    if (threadIdx.x == 0) atomicMax(gmax, red[0]);
}

// ---------------------------------------------------------------------------
// k4: exp(score - gmax), scatter-sum into denom[tgt][h]
// ---------------------------------------------------------------------------
__global__ void gat_exp_kernel(const int* __restrict__ tgt, float* __restrict__ attn,
                               const unsigned* __restrict__ gmax, float* __restrict__ denom)
{
    int tid = blockIdx.x * blockDim.x + threadIdx.x;
    if (tid >= N_EDGES * NUM_HEADS) return;
    const int e = tid >> 3, h = tid & 7;
    const float m = dec_f32(*gmax);
    const float es = expf(attn[tid] - m);
    attn[tid] = es;
    atomicAdd(&denom[tgt[e] * NUM_HEADS + h], es);
}

// ---------------------------------------------------------------------------
// k5: attn = exp/(denom+eps)/dist, scatter-sum into attn_sum[tgt][h]
// ---------------------------------------------------------------------------
__global__ void gat_norm1_kernel(const int* __restrict__ tgt, const float* __restrict__ dist,
                                 float* __restrict__ attn, const float* __restrict__ denom,
                                 float* __restrict__ asum)
{
    int tid = blockIdx.x * blockDim.x + threadIdx.x;
    if (tid >= N_EDGES * NUM_HEADS) return;
    const int e = tid >> 3, h = tid & 7;
    float a = attn[tid] / (denom[tgt[e] * NUM_HEADS + h] + GAT_EPS);
    a = a / dist[e];
    attn[tid] = a;
    atomicAdd(&asum[tgt[e] * NUM_HEADS + h], a);
}

// ---------------------------------------------------------------------------
// k6: final renorm + weighted scatter-aggregate into output
// ---------------------------------------------------------------------------
__global__ void gat_agg_kernel(const int* __restrict__ src, const int* __restrict__ tgt,
                               float* __restrict__ attn, const float* __restrict__ asum,
                               const float* __restrict__ value, float* __restrict__ out)
{
    int tid = blockIdx.x * blockDim.x + threadIdx.x;
    if (tid >= N_EDGES * NUM_HEADS) return;
    const int e = tid >> 3, h = tid & 7;
    const int t = tgt[e], s = src[e];
    const float a = attn[tid] / asum[t * NUM_HEADS + h];
    attn[tid] = a;                                   // final attn output
    const float4* vs = (const float4*)(value + (size_t)s * MODEL_DIM + h * D_V);
    float* od = out + (size_t)t * MODEL_DIM + h * D_V;
#pragma unroll
    for (int q = 0; q < 4; ++q) {
        float4 v = vs[q];
        atomicAdd(&od[q * 4 + 0], v.x * a);
        atomicAdd(&od[q * 4 + 1], v.y * a);
        atomicAdd(&od[q * 4 + 2], v.z * a);
        atomicAdd(&od[q * 4 + 3], v.w * a);
    }
}

// ---------------------------------------------------------------------------
extern "C" void kernel_launch(void* const* d_in, const int* in_sizes, int n_in,
                              void* d_out, int out_size, void* d_ws, size_t ws_size,
                              hipStream_t stream) {
    const float* inp   = (const float*)d_in[0];
    const int*   eidx  = (const int*)  d_in[1];   // [2, E]
    const float* dist  = (const float*)d_in[2];
    /* d_in[3] = deparc_edge, unused by reference */
    const float* Wv    = (const float*)d_in[4];
    const float* w_src = (const float*)d_in[5];
    const float* w_tgt = (const float*)d_in[6];
    const float* Wself = (const float*)d_in[7];
    const float* bias  = (const float*)d_in[8];

    const int* src = eidx;
    const int* tgt = eidx + N_EDGES;

    float* out  = (float*)d_out;                       // [N, 128]
    float* attn = out + (size_t)N_NODES * MODEL_DIM;   // [E, 8]

    float* value = (float*)d_ws;                               // [N,128]
    float* vsrc  = value + (size_t)N_NODES * MODEL_DIM;        // [N,8]
    float* vtgt  = vsrc  + (size_t)N_NODES * NUM_HEADS;
    float* denom = vtgt  + (size_t)N_NODES * NUM_HEADS;
    float* asum  = denom + (size_t)N_NODES * NUM_HEADS;
    unsigned* gmax = (unsigned*)(asum + (size_t)N_NODES * NUM_HEADS);

    const int NH  = N_NODES * NUM_HEADS;    // 400k
    const int EH  = N_EDGES * NUM_HEADS;    // 12.8M

    gat_zero_kernel<<<(NH + 255) / 256, 256, 0, stream>>>(denom, asum, gmax, NH);

    const int ntiles = (N_NODES + 15) / 16;            // 3125
    gat_gemm_wmma<<<(ntiles + 7) / 8, 256, 0, stream>>>(inp, Wv, Wself, bias, value, out);

    gat_proj_kernel<<<(NH + 255) / 256, 256, 0, stream>>>(value, w_src, w_tgt, vsrc, vtgt);

    gat_score_kernel<<<(EH + 255) / 256, 256, 0, stream>>>(src, tgt, vsrc, vtgt, attn, gmax);
    gat_exp_kernel  <<<(EH + 255) / 256, 256, 0, stream>>>(tgt, attn, gmax, denom);
    gat_norm1_kernel<<<(EH + 255) / 256, 256, 0, stream>>>(tgt, dist, attn, denom, asum);
    gat_agg_kernel  <<<(EH + 255) / 256, 256, 0, stream>>>(src, tgt, attn, asum, value, out);
}
